// ContinuousSpectrumMemory_69947837382657
// MI455X (gfx1250) — matrix operand order repaired and checked by reading, not compiled, not run on hardware
//
#include <hip/hip_runtime.h>

typedef __attribute__((ext_vector_type(16))) _Float16 v16h;
typedef __attribute__((ext_vector_type(8)))  _Float16 v8h;
typedef __attribute__((ext_vector_type(8)))  float    v8f;
typedef unsigned int u32x4 __attribute__((ext_vector_type(4)));
typedef int i32x8 __attribute__((ext_vector_type(8)));
typedef int i32x4 __attribute__((ext_vector_type(4)));
typedef __attribute__((address_space(3))) float lds_float_t;

namespace csm {
constexpr int B = 2, T = 4096, D = 1024, M = 256;
constexpr int ROWS = B * T;      // 8192 query rows
constexpr int LVLS = 13;         // tree levels: 4096,2048,...,1
constexpr int UPN  = T - 1;      // 4095 nodes in levels >= 1, per batch
}
using namespace csm;

// ---------------------------------------------------------------------------
// Elementwise conversion kernels
// ---------------------------------------------------------------------------
__global__ void prep_kernel(const float* __restrict__ mu, const float* __restrict__ lam,
                            _Float16* __restrict__ muh, _Float16* __restrict__ lvh) {
  size_t i = (size_t)blockIdx.x * blockDim.x + threadIdx.x;
  if (i >= (size_t)ROWS * D) return;
  muh[i] = (_Float16)mu[i];
  float la = lam[i];
  la = la < 1e-6f ? 1e-6f : la;
  lvh[i] = (_Float16)(-__logf(la));
}

__global__ void cvt_kernel(_Float16* __restrict__ dst, const float* __restrict__ src, int n) {
  int i = blockIdx.x * blockDim.x + threadIdx.x;
  if (i < n) dst[i] = (_Float16)src[i];
}

// Ws1 (256 x 2055) -> mu part (256x1024) and logvar part (256x1024), f16 packed
__global__ void ws1_split_kernel(_Float16* __restrict__ a, _Float16* __restrict__ b,
                                 const float* __restrict__ Ws1) {
  int i = blockIdx.x * blockDim.x + threadIdx.x;
  if (i >= 256 * 1024) return;
  int u = i >> 10, d = i & 1023;
  a[i] = (_Float16)Ws1[(size_t)u * 2055 + d];
  b[i] = (_Float16)Ws1[(size_t)u * 2055 + 1024 + d];
}

// ---------------------------------------------------------------------------
// Generic WMMA GEMM:  C(rows x N) = A(rows x K) * W(N x K)^T  [+ A2 * W2^T]
// f16 inputs, f32 accumulate.  Fragment layouts per CDNA5 ISA 7.12.2.
// ---------------------------------------------------------------------------
__device__ __forceinline__ v16h ld_a_frag(const _Float16* __restrict__ A, int K,
                                          int m, int kb, int hi) {
  // lane<16: K {kb..kb+7, kb+16..kb+23}; lane>=16: K {kb+8..15, kb+24..31}
  const _Float16* p = A + (size_t)m * K + kb + hi * 8;
  v8h g0 = *(const v8h*)p;
  v8h g1 = *(const v8h*)(p + 16);
  v16h r;
#pragma unroll
  for (int i = 0; i < 8; ++i) { r[i] = g0[i]; r[i + 8] = g1[i]; }
  return r;
}

__device__ __forceinline__ v16h ld_b_frag(const _Float16* __restrict__ W, int K,
                                          int n, int kb, int hi) {
  // lane<16: K {kb..kb+7, kb+16..kb+23}; lane>=16: K {kb+8..15, kb+24..31}; n = lane&15
  const _Float16* p = W + (size_t)n * K + kb + hi * 8;
  v8h g0 = *(const v8h*)p;
  v8h g1 = *(const v8h*)(p + 16);
  v16h r;
#pragma unroll
  for (int i = 0; i < 8; ++i) { r[i] = g0[i]; r[i + 8] = g1[i]; }
  return r;
}

__global__ void gemm_wmma_kernel(const _Float16* __restrict__ A,
                                 const _Float16* __restrict__ W,
                                 const _Float16* __restrict__ A2,
                                 const _Float16* __restrict__ W2,
                                 const float* __restrict__ bias,
                                 const float* __restrict__ gate,   // nullable: scale by sigmoid(*gate)
                                 float* __restrict__ C,            // nullable
                                 _Float16* __restrict__ Ch,        // nullable
                                 int rows, int N, int K) {
  int lane = threadIdx.x & 31;
  int tile = blockIdx.x * (blockDim.x >> 5) + (threadIdx.x >> 5);
  int ctiles = N >> 4;
  if (tile >= (rows >> 4) * ctiles) return;
  int tn = (tile % ctiles) << 4;
  int tm = (tile / ctiles) << 4;
  int hi = lane >> 4;
  int am = tm + (lane & 15);
  int bn = tn + (lane & 15);

  v8f c = {};
#pragma unroll 4
  for (int kb = 0; kb < K; kb += 32) {
    __builtin_prefetch(A + (size_t)am * K + kb + 128, 0, 0);
    __builtin_prefetch(W + (size_t)bn * K + kb + 128, 0, 0);
    v16h af = ld_a_frag(A, K, am, kb, hi);
    v16h bf = ld_b_frag(W, K, bn, kb, hi);
    c = __builtin_amdgcn_wmma_f32_16x16x32_f16(false, af, false, bf,
                                               (short)0, c, false, false);
  }
  if (A2 != nullptr) {
#pragma unroll 4
    for (int kb = 0; kb < K; kb += 32) {
      v16h af = ld_a_frag(A2, K, am, kb, hi);
      v16h bf = ld_b_frag(W2, K, bn, kb, hi);
      c = __builtin_amdgcn_wmma_f32_16x16x32_f16(false, af, false, bf,
                                                 (short)0, c, false, false);
    }
  }

  float scale = 1.0f;
  if (gate) scale = 1.0f / (1.0f + __expf(-gate[0]));
  int n = tn + (lane & 15);
  float bb = bias ? bias[n] : 0.0f;
  int mb = tm + (hi ? 8 : 0);
#pragma unroll
  for (int r = 0; r < 8; ++r) {
    float val = (c[r] + bb) * scale;
    size_t o = (size_t)(mb + r) * N + n;
    if (C)  C[o]  = val;
    if (Ch) Ch[o] = (_Float16)val;
  }
}

// ---------------------------------------------------------------------------
// Tree level reduction: weighted pairwise average of (k, v), weights summed.
// ---------------------------------------------------------------------------
__global__ void tree_reduce_kernel(const float* __restrict__ ks, const float* __restrict__ vs,
                                   const float* __restrict__ ws,
                                   float* __restrict__ kd, float* __restrict__ vd,
                                   float* __restrict__ wd,
                                   int n_out, int sbs, int dbs) {
  int idx = blockIdx.x * blockDim.x + threadIdx.x;
  int total = B * n_out * M;
  if (idx >= total) return;
  int m = idx & (M - 1);
  int j = (idx >> 8) % n_out;
  int b = idx / (n_out * M);
  size_t sn = (size_t)b * sbs + 2 * j;
  float w1 = ws[sn], w2 = ws[sn + 1];
  float tw = w1 + w2 + 1e-8f;
  float inv = 1.0f / tw;
  size_t s0 = sn * M + m;
  size_t d0 = ((size_t)b * dbs + j) * M + m;
  kd[d0] = (w1 * ks[s0] + w2 * ks[s0 + M]) * inv;
  vd[d0] = (w1 * vs[s0] + w2 * vs[s0 + M]) * inv;
  if (m == 0) wd[(size_t)b * dbs + j] = tw;
}

// ---------------------------------------------------------------------------
// Router head: s = sigmoid(Ws2 . silu(h + bs1 + pi*Ws1[:,2048:2055]) + bs2)
// ---------------------------------------------------------------------------
__global__ void router_kernel(const float* __restrict__ h, const float* __restrict__ pi,
                              const float* __restrict__ Ws1, const float* __restrict__ bs1,
                              const float* __restrict__ Ws2, const float* __restrict__ bs2,
                              float* __restrict__ sbuf) {
  int wid = (int)((blockIdx.x * blockDim.x + threadIdx.x) >> 5);
  int lane = threadIdx.x & 31;
  if (wid >= ROWS) return;
  float pv[7];
#pragma unroll
  for (int p = 0; p < 7; ++p) pv[p] = pi[(size_t)wid * 7 + p];
  float part = 0.0f;
#pragma unroll
  for (int iu = 0; iu < 8; ++iu) {
    int u = lane * 8 + iu;
    float hv = h[(size_t)wid * 256 + u] + bs1[u];
#pragma unroll
    for (int p = 0; p < 7; ++p) hv = fmaf(pv[p], Ws1[(size_t)u * 2055 + 2048 + p], hv);
    float sg = 1.0f / (1.0f + __expf(-hv));
    part = fmaf(hv * sg, Ws2[u], part);
  }
#pragma unroll
  for (int mm = 16; mm; mm >>= 1) part += __shfl_xor(part, mm, 32);
  if (lane == 0) sbuf[wid] = 1.0f / (1.0f + __expf(-(part + bs2[0])));
}

// ---------------------------------------------------------------------------
// Scratch attention: softmax over S=8 fixed slots, context per row.
// ---------------------------------------------------------------------------
__global__ void scratch_kernel(const float* __restrict__ sq, const float* __restrict__ scr,
                               float* __restrict__ sctx) {
  int wid = (int)((blockIdx.x * blockDim.x + threadIdx.x) >> 5);
  int lane = threadIdx.x & 31;
  if (wid >= ROWS) return;
  float qv[8];
  const float* qp = sq + (size_t)wid * 256 + lane * 8;
#pragma unroll
  for (int i = 0; i < 8; ++i) qv[i] = qp[i];
  float sc[8];
#pragma unroll
  for (int slot = 0; slot < 8; ++slot) {
    const float* kp = scr + slot * 256 + lane * 8;
    float d = 0.0f;
#pragma unroll
    for (int i = 0; i < 8; ++i) d = fmaf(qv[i], kp[i], d);
#pragma unroll
    for (int mm = 16; mm; mm >>= 1) d += __shfl_xor(d, mm, 32);
    sc[slot] = d * 0.0625f;   // 1/sqrt(256)
  }
  float mx = sc[0];
#pragma unroll
  for (int i = 1; i < 8; ++i) mx = fmaxf(mx, sc[i]);
  float es = 0.0f;
#pragma unroll
  for (int i = 0; i < 8; ++i) { sc[i] = __expf(sc[i] - mx); es += sc[i]; }
  float inv = 1.0f / es;
  float outc[8] = {0, 0, 0, 0, 0, 0, 0, 0};
#pragma unroll
  for (int slot = 0; slot < 8; ++slot) {
    float p = sc[slot] * inv;
    const float* vp = scr + slot * 256 + lane * 8;
#pragma unroll
    for (int i = 0; i < 8; ++i) outc[i] = fmaf(p, vp[i], outc[i]);
  }
  float* op = sctx + (size_t)wid * 256 + lane * 8;
#pragma unroll
  for (int i = 0; i < 8; ++i) op[i] = outc[i];
}

// ---------------------------------------------------------------------------
// TDM: DMA `rows` key rows (256 f32 each, contiguous) from global into LDS.
// 2D descriptor per CDNA5 ISA 8.3/8.4: tensor_dim0 = tile_dim0 = 256 elems,
// data_size = 4B, tensor_dim0_stride = 256, tile_dim1 = tensor_dim1 = rows.
// Issued by one wave (TDM ignores EXEC); tracked with TENSORcnt.
// ---------------------------------------------------------------------------
__device__ __forceinline__ void tdm_load_keys(const float* gsrc, float* ldst, int rows) {
  unsigned long long ga = (unsigned long long)(size_t)gsrc;
  unsigned lofs = (unsigned)(size_t)(lds_float_t*)ldst;  // LDS byte offset
  u32x4 g0;
  g0[0] = 1u;                                            // count=1, user descriptor
  g0[1] = lofs;                                          // lds_addr
  g0[2] = (unsigned)(ga & 0xffffffffu);                  // global_addr[31:0]
  g0[3] = (unsigned)((ga >> 32) & 0x01ffffffu) | (2u << 30);  // addr[56:32] | type=2
  i32x8 g1;
  g1[0] = (int)(2u << 16);                               // data_size=2 (4 bytes)
  g1[1] = (int)((256u & 0xffffu) << 16);                 // tensor_dim0[15:0]
  g1[2] = (int)(((unsigned)rows & 0xffffu) << 16);       // tensor_dim0[31:16]=0 | tensor_dim1[15:0]
  g1[3] = (int)(256u << 16);                             // tensor_dim1[31:16]=0 | tile_dim0=256
  g1[4] = rows;                                          // tile_dim1=rows, tile_dim2=0
  g1[5] = 256;                                           // tensor_dim0_stride[31:0]
  g1[6] = 0;                                             // stride hi / dim1_stride lo
  g1[7] = 0;
  i32x4 z4 = {0, 0, 0, 0};
#if defined(__clang_major__) && (__clang_major__ >= 23)
  i32x8 z8 = {0, 0, 0, 0, 0, 0, 0, 0};
  __builtin_amdgcn_tensor_load_to_lds(g0, g1, z4, z4, z8, 0);
#else
  __builtin_amdgcn_tensor_load_to_lds(g0, g1, z4, z4, 0);
#endif
}

// ---------------------------------------------------------------------------
// Beam retrieval over all 13 tree levels + kappa blend + scratch blend.
// Block = 4 waves = 4 consecutive queries (same batch); key tiles staged into
// LDS by the Tensor Data Mover (wave 0 issues, s_wait_tensorcnt fences).
// ---------------------------------------------------------------------------
__global__ void retrieve_kernel(const float* __restrict__ q,
                                const float* __restrict__ k0, const float* __restrict__ v0,
                                const float* __restrict__ kup, const float* __restrict__ vup,
                                const float* __restrict__ sbuf, const float* __restrict__ sctx,
                                _Float16* __restrict__ Ch) {
  __shared__ __align__(16) float kst[8 * 256];
  int wv = threadIdx.x >> 5;
  int lane = threadIdx.x & 31;
  int wid = blockIdx.x * 4 + wv;            // ROWS % 4 == 0: always valid
  int b = wid >> 12;
  int t = wid & (T - 1);
  int tLast = ((blockIdx.x * 4 + 3) & (T - 1));  // block never straddles batches

  float ql[8];
  const float* qp = q + (size_t)wid * 256 + lane * 8;
#pragma unroll
  for (int i = 0; i < 8; ++i) ql[i] = qp[i];

  float s = sbuf[wid];
  float kap[LVLS], ksum = 0.0f;
  float cen = s * (float)(LVLS - 1);
#pragma unroll
  for (int l = 0; l < LVLS; ++l) {
    float dl = (float)l - cen;
    kap[l] = __expf(-0.5f * dl * dl);
    ksum += kap[l];
  }
  float kinv = 1.0f / fmaxf(ksum, 1e-8f);

  float acc[8] = {0, 0, 0, 0, 0, 0, 0, 0};

  for (int l = 0; l < LVLS; ++l) {
    int n = T >> l;
    const float* kb;
    const float* vb;
    if (l == 0) {
      kb = k0 + (size_t)b * T * M;
      vb = v0 + (size_t)b * T * M;
    } else {
      int off = T - ((2 * T) >> l);  // 0, 2048, 3072, ... 4094
      kb = kup + ((size_t)b * UPN + off) * M;
      vb = vup + ((size_t)b * UPN + off) * M;
    }
    int jmax  = t < n - 1 ? t : n - 1;        // causal: node index <= t
    int jmaxB = tLast < n - 1 ? tLast : n - 1;

    float s0 = -1e30f, s1 = -1e30f, s2 = -1e30f, s3 = -1e30f;
    int i0 = 0, i1 = 0, i2 = 0, i3 = 0;

    for (int base = 0; base <= jmaxB; base += 8) {
      int cnt = jmaxB - base + 1;
      if (cnt > 8) cnt = 8;
      __syncthreads();                        // all waves done with prev tile
      if (wv == 0) {
        tdm_load_keys(kb + (size_t)base * 256, kst, cnt);
        __builtin_amdgcn_s_wait_tensorcnt(0); // fence DMA in issuing wave
      }
      __syncthreads();                        // tile visible to all waves
      int jend = jmax < base + cnt - 1 ? jmax : base + cnt - 1;
      for (int j = base; j <= jend; ++j) {
        const float4* kp = (const float4*)(kst + (j - base) * 256 + lane * 8);
        float4 x = kp[0], y = kp[1];
        float d = ql[0] * x.x + ql[1] * x.y + ql[2] * x.z + ql[3] * x.w +
                  ql[4] * y.x + ql[5] * y.y + ql[6] * y.z + ql[7] * y.w;
#pragma unroll
        for (int mm = 16; mm; mm >>= 1) d += __shfl_xor(d, mm, 32);
        d *= 0.0625f;  // 1/sqrt(M)
        if (d > s2) {
          if (d > s0)      { s3 = s2; i3 = i2; s2 = s1; i2 = i1; s1 = s0; i1 = i0; s0 = d; i0 = j; }
          else if (d > s1) { s3 = s2; i3 = i2; s2 = s1; i2 = i1; s1 = d; i1 = j; }
          else             { s3 = s2; i3 = i2; s2 = d; i2 = j; }
        } else if (d > s3) { s3 = d; i3 = j; }
      }
    }

    // softmax over the (<=4) beam entries; empty slots stayed at -1e30 -> 0
    float e0 = 1.0f;
    float e1 = __expf(s1 - s0), e2 = __expf(s2 - s0), e3 = __expf(s3 - s0);
    float wl = kap[l] * kinv / (e0 + e1 + e2 + e3);
    const float* v0p = vb + (size_t)i0 * 256 + lane * 8;
    const float* v1p = vb + (size_t)i1 * 256 + lane * 8;
    const float* v2p = vb + (size_t)i2 * 256 + lane * 8;
    const float* v3p = vb + (size_t)i3 * 256 + lane * 8;
#pragma unroll
    for (int i = 0; i < 8; ++i) {
      float lv = e0 * v0p[i] + e1 * v1p[i] + e2 * v2p[i] + e3 * v3p[i];
      acc[i] = fmaf(wl, lv, acc[i]);
    }
  }

  const float* sp = sctx + (size_t)wid * 256 + lane * 8;
  _Float16* op = Ch + (size_t)wid * 256 + lane * 8;
#pragma unroll
  for (int i = 0; i < 8; ++i) {
    float comb = (1.0f - s) * acc[i] + s * sp[i];
    op[i] = (_Float16)comb;
  }
}

// ---------------------------------------------------------------------------
// Host orchestration
// ---------------------------------------------------------------------------
extern "C" void kernel_launch(void* const* d_in, const int* in_sizes, int n_in,
                              void* d_out, int out_size, void* d_ws, size_t ws_size,
                              hipStream_t stream) {
  (void)in_sizes; (void)n_in; (void)out_size; (void)ws_size;
  const float* mu   = (const float*)d_in[0];
  const float* lam  = (const float*)d_in[1];
  const float* pi   = (const float*)d_in[2];
  const float* piw  = (const float*)d_in[3];
  const float* Wq   = (const float*)d_in[4];
  const float* bq   = (const float*)d_in[5];
  const float* Wk   = (const float*)d_in[6];
  const float* bk   = (const float*)d_in[7];
  const float* Wv   = (const float*)d_in[8];
  const float* bv   = (const float*)d_in[9];
  const float* Wo   = (const float*)d_in[10];
  const float* bo   = (const float*)d_in[11];
  const float* Ws1  = (const float*)d_in[12];
  const float* bs1  = (const float*)d_in[13];
  const float* Ws2  = (const float*)d_in[14];
  const float* bs2  = (const float*)d_in[15];
  const float* Wsr  = (const float*)d_in[16];
  const float* bsr  = (const float*)d_in[17];
  const float* scri = (const float*)d_in[18];
  const float* gate = (const float*)d_in[19];
  float* out = (float*)d_out;

  char* wp = (char*)d_ws;
  auto alloc = [&](size_t bytes) -> char* {
    char* p = wp; wp += (bytes + 255) & ~(size_t)255; return p;
  };

  _Float16* muh  = (_Float16*)alloc((size_t)ROWS * D * 2);
  _Float16* lvh  = (_Float16*)alloc((size_t)ROWS * D * 2);
  _Float16* Wqh  = (_Float16*)alloc((size_t)M * D * 2);
  _Float16* Wkh  = (_Float16*)alloc((size_t)M * D * 2);
  _Float16* Wvh  = (_Float16*)alloc((size_t)M * D * 2);
  _Float16* Woh  = (_Float16*)alloc((size_t)D * M * 2);
  _Float16* Wsrh = (_Float16*)alloc((size_t)M * M * 2);
  _Float16* W1a  = (_Float16*)alloc((size_t)256 * 1024 * 2);
  _Float16* W1b  = (_Float16*)alloc((size_t)256 * 1024 * 2);
  float*    qf   = (float*)alloc((size_t)ROWS * M * 4);
  _Float16* qh   = (_Float16*)alloc((size_t)ROWS * M * 2);
  float*    k0   = (float*)alloc((size_t)ROWS * M * 4);
  float*    v0   = (float*)alloc((size_t)ROWS * M * 4);
  float*    kup  = (float*)alloc((size_t)B * UPN * M * 4);
  float*    vup  = (float*)alloc((size_t)B * UPN * M * 4);
  float*    wup  = (float*)alloc((size_t)B * UPN * 4);
  float*    hbuf = (float*)alloc((size_t)ROWS * 256 * 4);
  float*    sbuf = (float*)alloc((size_t)ROWS * 4);
  float*    sqf  = (float*)alloc((size_t)ROWS * M * 4);
  float*    sctx = (float*)alloc((size_t)ROWS * M * 4);
  _Float16* cmbh = (_Float16*)alloc((size_t)ROWS * M * 2);

  // 1) activation / weight conversion
  {
    int n = ROWS * D;
    prep_kernel<<<(n + 255) / 256, 256, 0, stream>>>(mu, lam, muh, lvh);
    cvt_kernel<<<(M * D + 255) / 256, 256, 0, stream>>>(Wqh, Wq, M * D);
    cvt_kernel<<<(M * D + 255) / 256, 256, 0, stream>>>(Wkh, Wk, M * D);
    cvt_kernel<<<(M * D + 255) / 256, 256, 0, stream>>>(Wvh, Wv, M * D);
    cvt_kernel<<<(D * M + 255) / 256, 256, 0, stream>>>(Woh, Wo, D * M);
    cvt_kernel<<<(M * M + 255) / 256, 256, 0, stream>>>(Wsrh, Wsr, M * M);
    ws1_split_kernel<<<(256 * 1024) / 256, 256, 0, stream>>>(W1a, W1b, Ws1);
  }

  // 2) projections (WMMA): q (f32 + f16), k, v, router hidden
  {
    dim3 blk(128);
    int tiles = (ROWS / 16) * (M / 16);  // 8192
    gemm_wmma_kernel<<<tiles / 4, blk, 0, stream>>>(muh, Wqh, nullptr, nullptr, bq, nullptr,
                                                    qf, qh, ROWS, M, D);
    gemm_wmma_kernel<<<tiles / 4, blk, 0, stream>>>(muh, Wkh, nullptr, nullptr, bk, nullptr,
                                                    k0, nullptr, ROWS, M, D);
    gemm_wmma_kernel<<<tiles / 4, blk, 0, stream>>>(muh, Wvh, nullptr, nullptr, bv, nullptr,
                                                    v0, nullptr, ROWS, M, D);
    gemm_wmma_kernel<<<tiles / 4, blk, 0, stream>>>(muh, W1a, lvh, W1b, nullptr, nullptr,
                                                    hbuf, nullptr, ROWS, 256, 1024);
  }

  // 3) router scalar s
  router_kernel<<<ROWS / 4, 128, 0, stream>>>(hbuf, pi, Ws1, bs1, Ws2, bs2, sbuf);

  // 4) tree build (levels 1..12); level-0 weights are pi_write directly
  {
    const float* ks = k0; const float* vs = v0; const float* ws = piw;
    int sbs = T;
    for (int l = 1; l < LVLS; ++l) {
      int n_out = T >> l;
      int off_d = T - ((2 * T) >> l);
      float* kd = kup + (size_t)off_d * M;
      float* vd = vup + (size_t)off_d * M;
      float* wd = wup + off_d;
      int total = B * n_out * M;
      tree_reduce_kernel<<<(total + 255) / 256, 256, 0, stream>>>(ks, vs, ws, kd, vd, wd,
                                                                  n_out, sbs, UPN);
      ks = kd; vs = vd; ws = wd; sbs = UPN;
    }
  }

  // 5) scratch attention: sq = q@Wsr^T + bsr (WMMA), then softmax over S=8
  {
    dim3 blk(128);
    int tiles = (ROWS / 16) * (M / 16);
    gemm_wmma_kernel<<<tiles / 4, blk, 0, stream>>>(qh, Wsrh, nullptr, nullptr, bsr, nullptr,
                                                    sqf, nullptr, ROWS, M, M);
    scratch_kernel<<<ROWS / 4, 128, 0, stream>>>(sqf, scri, sctx);
  }

  // 6) beam retrieval over all levels + kappa/scratch blend -> combined (f16)
  retrieve_kernel<<<ROWS / 4, 128, 0, stream>>>(qf, k0, v0, kup, vup, sbuf, sctx, cmbh);

  // 7) output projection (WMMA) with sigmoid(gate) scale
  {
    dim3 blk(128);
    int tiles = (ROWS / 16) * (D / 16);  // 32768
    gemm_wmma_kernel<<<tiles / 4, blk, 0, stream>>>(cmbh, Woh, nullptr, nullptr, bo, gate,
                                                    out, nullptr, ROWS, D, M);
  }
}
